// RobustConv_51505247814292
// MI455X (gfx1250) — compile-verified, matrix operand order checked by reading, not run on hardware
//
#include <hip/hip_runtime.h>

// ---------------------------------------------------------------------------
// RobustConv for MI455X (gfx1250, wave32, WMMA).
//   mean = relu(x@Wm + bm); var = relu(x@Wv + bv); att = exp(-var);
//   mean *= att; var *= att*att;  then GCN-normalized scatter w/ self loops.
// GEMM: v_wmma_f32_16x16x32_bf16; W pre-converted once to bf16 (K-major),
// staged into LDS in 32KB tiles; B fragments double-buffered so ds_load_b128
// overlaps WMMA issue; x streamed with TH_LOAD_NT (no reuse, keep L2 clean).
// Intermediates stored bf16 to halve scatter bandwidth (dominant phase).
// ---------------------------------------------------------------------------

#define IN_C  256
#define OUT_C 128

typedef __attribute__((ext_vector_type(16))) __bf16        v16bf;
typedef __attribute__((ext_vector_type(2)))  __bf16        v2bf;
typedef __attribute__((ext_vector_type(8)))  float         v8f;
typedef __attribute__((ext_vector_type(4)))  float         f32x4;
typedef __attribute__((ext_vector_type(4)))  unsigned int  u32x4;
typedef __attribute__((ext_vector_type(4)))  unsigned short u16x4;

union FragB { v16bf v; u32x4 q[2]; };

__device__ __forceinline__ unsigned pack2_bf16(float a, float b) {
    v2bf v;                       // native convert: v_cvt_pk_bf16_f32
    v[0] = (__bf16)a;
    v[1] = (__bf16)b;
    unsigned r;
    __builtin_memcpy(&r, &v, 4);
    return r;
}
__device__ __forceinline__ float bf2f(unsigned short h) {
    return __uint_as_float(((unsigned)h) << 16);
}

// -------------------------------- utility kernels ---------------------------

__global__ void zero_kernel(f32x4* __restrict__ p, long long n4) {
    long long i = (long long)blockIdx.x * blockDim.x + threadIdx.x;
    long long stride = (long long)gridDim.x * blockDim.x;
    f32x4 z = {0.f, 0.f, 0.f, 0.f};
    for (; i < n4; i += stride) p[i] = z;
}

__global__ void fill1_kernel(float* __restrict__ deg, int n) {
    int i = blockIdx.x * blockDim.x + threadIdx.x;
    if (i < n) deg[i] = 1.0f;   // self-loop contribution
}

__global__ void degree_kernel(const int* __restrict__ cols, float* __restrict__ deg, int M) {
    int e = blockIdx.x * blockDim.x + threadIdx.x;
    if (e < M) atomicAdd(&deg[cols[e]], 1.0f);
}

__global__ void rsqrt_kernel(float* __restrict__ deg, int n) {
    int i = blockIdx.x * blockDim.x + threadIdx.x;
    if (i < n) deg[i] = rsqrtf(deg[i]);   // deg >= 1 always (self loops)
}

// One-time weight prep: fp32 W[k][n] -> bf16 Wt[mat][n][k] (K-major, so the
// per-block LDS fill is contiguous b128 copies with zero converts).
__global__ __launch_bounds__(256)
void wprep_kernel(const float* __restrict__ Wm, const float* __restrict__ Wv,
                  unsigned short* __restrict__ Wt)
{
    const int t   = threadIdx.x;      // single block of 256: one (mat,n) each
    const int mat = t >> 7;
    const int n   = t & 127;
    const float* Wsrc = mat ? Wv : Wm;
    unsigned* dst = (unsigned*)&Wt[(size_t)((mat * OUT_C) + n) * IN_C];
    #pragma unroll 4
    for (int k = 0; k < IN_C; k += 2)
        dst[k >> 1] = pack2_bf16(Wsrc[(size_t)k       * OUT_C + n],
                                 Wsrc[(size_t)(k + 1) * OUT_C + n]);
}

// -------------------------- fused WMMA GEMM + epilogue -----------------------
// Block = 256 threads = 8 waves. Each wave: 16 rows x 128 cols, both matrices.
// K tiled in 64-wide LDS chunks (both W matrices, bf16, 32KB).

__global__ __launch_bounds__(256)
void gemm_epilogue_kernel(const float* __restrict__ x,
                          const unsigned short* __restrict__ Wt,
                          const float* __restrict__ bm,
                          const float* __restrict__ bv,
                          unsigned short* __restrict__ mean_bf,
                          unsigned short* __restrict__ var_bf,
                          int N)
{
    // sW[mat][n][kk] : 2 * 128 * 64 bf16 = 32 KB, K-major per column for b128 reads
    __shared__ __align__(16) unsigned short sW[2 * OUT_C * 64];

    const int tid   = threadIdx.x;
    const int wave  = tid >> 5;
    const int lane  = tid & 31;
    const int khalf = lane >> 4;      // 0: K 0-7/16-23, 1: K 8-15/24-31
    const int l16   = lane & 15;
    const int rowBase = blockIdx.x * 128 + wave * 16;

    v8f accm[8] = {};
    v8f accv[8] = {};

    // A-matrix row for this lane (clamped: EXEC must stay all-ones for WMMA)
    const int ar = min(rowBase + l16, N - 1);
    const float* xr = x + (size_t)ar * IN_C;

    for (int kc = 0; kc < IN_C; kc += 64) {
        // ---- LDS fill: contiguous bf16 copy, one (mat,n) column per thread ----
        {
            const u32x4* src = (const u32x4*)&Wt[(size_t)tid * IN_C + kc]; // tid==(mat,n)
            u32x4*       dst = (u32x4*)&sW[tid * 64];
            #pragma unroll
            for (int i = 0; i < 8; ++i) dst[i] = src[i];
        }
        __syncthreads();

        #pragma unroll
        for (int s = 0; s < 2; ++s) {
            const int k0 = kc + s * 32;
            // ---- A fragment: K groups [khalf*8 .. +7] and [16+khalf*8 .. +7]
            union { v16bf v; unsigned u[8]; } a;
            {
                const float* g = xr + k0 + khalf * 8;
                f32x4 f0 = __builtin_nontemporal_load((const f32x4*)(g));
                f32x4 f1 = __builtin_nontemporal_load((const f32x4*)(g + 4));
                f32x4 f2 = __builtin_nontemporal_load((const f32x4*)(g + 16));
                f32x4 f3 = __builtin_nontemporal_load((const f32x4*)(g + 20));
                a.u[0] = pack2_bf16(f0.x, f0.y);
                a.u[1] = pack2_bf16(f0.z, f0.w);
                a.u[2] = pack2_bf16(f1.x, f1.y);
                a.u[3] = pack2_bf16(f1.z, f1.w);
                a.u[4] = pack2_bf16(f2.x, f2.y);
                a.u[5] = pack2_bf16(f2.z, f2.w);
                a.u[6] = pack2_bf16(f3.x, f3.y);
                a.u[7] = pack2_bf16(f3.z, f3.w);
            }
            const int kofs = s * 32 + khalf * 8;

            // ---- double-buffered B fragments: prefetch tile nt+1 while
            //      tile nt's two WMMAs issue (wait depth stays >= 4) ----
            FragB fm[2], fv[2];
            {
                const unsigned short* pm = &sW[(l16) * 64 + kofs];
                const unsigned short* pv = pm + OUT_C * 64;
                fm[0].q[0] = *(const u32x4*)(pm);
                fm[0].q[1] = *(const u32x4*)(pm + 16);
                fv[0].q[0] = *(const u32x4*)(pv);
                fv[0].q[1] = *(const u32x4*)(pv + 16);
            }
            #pragma unroll
            for (int nt = 0; nt < 8; ++nt) {
                if (nt < 7) {
                    const int b = (nt + 1) & 1;
                    const unsigned short* pm = &sW[((nt + 1) * 16 + l16) * 64 + kofs];
                    const unsigned short* pv = pm + OUT_C * 64;
                    fm[b].q[0] = *(const u32x4*)(pm);
                    fm[b].q[1] = *(const u32x4*)(pm + 16);
                    fv[b].q[0] = *(const u32x4*)(pv);
                    fv[b].q[1] = *(const u32x4*)(pv + 16);
                }
                accm[nt] = __builtin_amdgcn_wmma_f32_16x16x32_bf16(
                    false, a.v, false, fm[nt & 1].v, (short)0, accm[nt], false, false);
                accv[nt] = __builtin_amdgcn_wmma_f32_16x16x32_bf16(
                    false, a.v, false, fv[nt & 1].v, (short)0, accv[nt], false, false);
            }
        }
        __syncthreads();
    }

    // ---- fused epilogue: bias, relu, attention scaling, bf16 store ----
    #pragma unroll
    for (int nt = 0; nt < 8; ++nt) {
        const int col = nt * 16 + l16;
        const float biasm = bm[col];
        const float biasv = bv[col];
        #pragma unroll
        for (int v = 0; v < 8; ++v) {
            const int row = rowBase + khalf * 8 + v;   // C/D layout: VGPR v -> M=v(+8)
            if (row < N) {
                float me = accm[nt][v] + biasm;
                float va = accv[nt][v] + biasv;
                me = fmaxf(me, 0.0f);
                va = fmaxf(va, 0.0f);
                float att = __expf(-va);               // GAMMA = 1.0
                me *= att;
                va *= att * att;
                __bf16 meb = (__bf16)me;
                __bf16 vab = (__bf16)va;
                unsigned short mes, vas;
                __builtin_memcpy(&mes, &meb, 2);
                __builtin_memcpy(&vas, &vab, 2);
                mean_bf[(size_t)row * OUT_C + col] = mes;
                var_bf [(size_t)row * OUT_C + col] = vas;
            }
        }
    }
}

// ------------------------------- edge scatter --------------------------------
// One wave per (edge | self-loop); lane handles 4 mean + 4 var channels.

__global__ __launch_bounds__(256)
void scatter_kernel(const int* __restrict__ rows, const int* __restrict__ cols,
                    const float* __restrict__ dinv,
                    const unsigned short* __restrict__ mean_bf,
                    const unsigned short* __restrict__ var_bf,
                    float* __restrict__ mean_out, float* __restrict__ var_out,
                    int M, int N)
{
    const long long wid = ((long long)blockIdx.x * blockDim.x + threadIdx.x) >> 5;
    const int lane = threadIdx.x & 31;
    if (wid >= (long long)M + N) return;

    int r, c;
    if (wid < M) { r = rows[wid]; c = cols[wid]; }
    else         { r = c = (int)(wid - M); }       // self loop

    const float norm  = dinv[r] * dinv[c];
    const float norm2 = norm * norm;
    const int ch = lane * 4;

    u16x4 mb = *(const u16x4*)&mean_bf[(size_t)c * OUT_C + ch];
    u16x4 vb = *(const u16x4*)&var_bf [(size_t)c * OUT_C + ch];

    float* mo = &mean_out[(size_t)r * OUT_C + ch];
    float* vo = &var_out [(size_t)r * OUT_C + ch];
    atomicAdd(mo + 0, norm  * bf2f(mb.x));
    atomicAdd(mo + 1, norm  * bf2f(mb.y));
    atomicAdd(mo + 2, norm  * bf2f(mb.z));
    atomicAdd(mo + 3, norm  * bf2f(mb.w));
    atomicAdd(vo + 0, norm2 * bf2f(vb.x));
    atomicAdd(vo + 1, norm2 * bf2f(vb.y));
    atomicAdd(vo + 2, norm2 * bf2f(vb.z));
    atomicAdd(vo + 3, norm2 * bf2f(vb.w));
}

// --------------------------------- launch ------------------------------------

extern "C" void kernel_launch(void* const* d_in, const int* in_sizes, int n_in,
                              void* d_out, int out_size, void* d_ws, size_t ws_size,
                              hipStream_t stream)
{
    const float* x  = (const float*)d_in[0];
    const float* Wm = (const float*)d_in[1];
    const float* Wv = (const float*)d_in[2];
    const float* bm = (const float*)d_in[3];
    const float* bv = (const float*)d_in[4];
    const int*   ei = (const int*)d_in[5];

    const int N = in_sizes[0] / IN_C;
    const int M = in_sizes[5] / 2;
    const int* rows = ei;
    const int* cols = ei + M;

    // workspace: [deg: N f32][Wt: 2*128*256 bf16][mean: N*128 bf16][var: N*128 bf16]
    char* ws = (char*)d_ws;
    float* deg = (float*)ws;
    size_t off = (((size_t)N * 4) + 255) & ~(size_t)255;
    unsigned short* Wt = (unsigned short*)(ws + off);
    off += (size_t)2 * OUT_C * IN_C * 2;
    unsigned short* mean_bf = (unsigned short*)(ws + off);
    unsigned short* var_bf  = (unsigned short*)(ws + off + (size_t)N * OUT_C * 2);

    float* mean_out = (float*)d_out;
    float* var_out  = mean_out + (size_t)N * OUT_C;

    // 1) zero outputs (out_size = 2*N*128, divisible by 4)
    long long n4 = (long long)out_size / 4;
    zero_kernel<<<4096, 256, 0, stream>>>((f32x4*)d_out, n4);

    // 2-4) degree -> dinv, plus one-time bf16 weight transpose
    fill1_kernel <<<(N + 255) / 256, 256, 0, stream>>>(deg, N);
    degree_kernel<<<(M + 255) / 256, 256, 0, stream>>>(cols, deg, M);
    rsqrt_kernel <<<(N + 255) / 256, 256, 0, stream>>>(deg, N);
    wprep_kernel <<<1, 256, 0, stream>>>(Wm, Wv, Wt);

    // 5) fused dual GEMM + epilogue (128 rows per block)
    gemm_epilogue_kernel<<<(N + 127) / 128, 256, 0, stream>>>(
        x, Wt, bm, bv, mean_bf, var_bf, N);

    // 6) normalized propagation: one wave per edge (+N self loops)
    long long nwaves = (long long)M + N;
    int sblocks = (int)((nwaves + 7) / 8);   // 8 waves per 256-thread block
    scatter_kernel<<<sblocks, 256, 0, stream>>>(
        rows, cols, deg, mean_bf, var_bf, mean_out, var_out, M, N);
}
